// TiedRowAttention_61117384622519
// MI455X (gfx1250) — matrix operand order repaired and checked by reading, not compiled, not run on hardware
//
#include <hip/hip_runtime.h>
#include <hip/hip_bf16.h>
#include <math.h>

#define D_MODEL 768
#define NHEAD   12
#define HEAD_DIM 64
#define SEQ     4096
#define NQKV    2304   // 3 * D_MODEL
#define SCALE_Q 0.03608439182435161f  // 1/sqrt(768)

typedef _Float16 half_t;
typedef __attribute__((ext_vector_type(16))) _Float16 v16h;
typedef __attribute__((ext_vector_type(8)))  _Float16 h8;
typedef __attribute__((ext_vector_type(8)))  float    v8f;

static __device__ __forceinline__ v16h make_frag(h8 lo, h8 hi) {
    v16h r;
#pragma unroll
    for (int i = 0; i < 8; ++i) { r[i] = lo[i]; r[i + 8] = hi[i]; }
    return r;
}

// A-fragment (16x32 f16). Caller passes the row base for row = (lane & 15).
// Lanes 0-15 hold K [0..7] and [16..23]; lanes 16-31 hold K [8..15] and [24..31].
static __device__ __forceinline__ v16h load_a_frag(const half_t* __restrict__ base, int lane) {
    int ko = (lane >> 4) * 8;
    h8 lo = *(const h8*)(base + ko);
    h8 hi = *(const h8*)(base + 16 + ko);
    return make_frag(lo, hi);
}

// B-fragment (32x16 f16). Caller passes pointer to the 32-element K-run for
// column n = (lane & 15). Lanes 0-15 hold K [0..7] and [16..23]; lanes 16-31 hold K [8..15] and [24..31].
static __device__ __forceinline__ v16h load_b_frag(const half_t* __restrict__ base, int lane) {
    int kb = (lane >> 4) * 8;
    h8 lo = *(const h8*)(base + kb);
    h8 hi = *(const h8*)(base + 16 + kb);
    return make_frag(lo, hi);
}

static __device__ __forceinline__ v8f wmma_f16(v16h a, v16h b, v8f c) {
    return __builtin_amdgcn_wmma_f32_16x16x32_f16(false, a, false, b, (short)0, c, false, false);
}

// ---------------- conversion kernels ----------------

__global__ void tra_cvt_f32_to_f16(const float* __restrict__ src, half_t* __restrict__ dst, int n) {
    int i = blockIdx.x * blockDim.x + threadIdx.x;
    if (i < n) dst[i] = (half_t)src[i];
}

// src[rows][cols] (f32) -> dst[cols][rows] (f16)
__global__ void tra_cvt_transpose(const float* __restrict__ src, half_t* __restrict__ dst,
                                  int rows, int cols) {
    int i = blockIdx.x * blockDim.x + threadIdx.x;
    if (i < rows * cols) {
        int r = i / cols, c = i - r * cols;
        dst[(size_t)c * rows + r] = (half_t)src[i];
    }
}

// ---------------- QKV GEMM: x16[4096][768] @ WqkvT[2304][768]^T + b ----------------
// One wave computes a 16x64 output strip, software-pipelined k-loop: loads for
// chunk k+32 issue before the WMMAs on chunk k so the VMEM latency is hidden.

__global__ void tra_qkv_gemm(const half_t* __restrict__ x16, const half_t* __restrict__ WqkvT,
                             const float* __restrict__ bqkv,
                             half_t* __restrict__ Q, half_t* __restrict__ K,
                             half_t* __restrict__ Vt) {
    int lane = threadIdx.x & 31, wave = threadIdx.x >> 5;
    int tile  = blockIdx.x * 8 + wave;
    int tileN = tile % (NQKV / 64);
    int tileM = tile / (NQKV / 64);

    const half_t* arow = x16   + (size_t)(tileM * 16 + (lane & 15)) * D_MODEL;
    const half_t* brow = WqkvT + (size_t)(tileN * 64 + (lane & 15)) * D_MODEL;

    v8f acc[4] = {{}, {}, {}, {}};

    v16h a0 = load_a_frag(arow, lane);
    v16h b0[4];
#pragma unroll
    for (int s = 0; s < 4; ++s) b0[s] = load_b_frag(brow + (size_t)s * 16 * D_MODEL, lane);

#pragma unroll 2
    for (int k0 = 0; k0 < D_MODEL; k0 += 32) {
        int k1 = (k0 + 32 == D_MODEL) ? 0 : k0 + 32;   // branchless tail (redundant reload)
        v16h a1 = load_a_frag(arow + k1, lane);
        v16h b1[4];
#pragma unroll
        for (int s = 0; s < 4; ++s)
            b1[s] = load_b_frag(brow + (size_t)s * 16 * D_MODEL + k1, lane);
#pragma unroll
        for (int s = 0; s < 4; ++s) acc[s] = wmma_f16(a0, b0[s], acc[s]);
        a0 = a1;
#pragma unroll
        for (int s = 0; s < 4; ++s) b0[s] = b1[s];
    }

#pragma unroll
    for (int s = 0; s < 4; ++s) {
        int   n    = tileN * 64 + s * 16 + (lane & 15);
        float bias = bqkv[n];
        int t  = n / D_MODEL;
        int rm = n - t * D_MODEL;
        int hh = rm >> 6, d = rm & 63;
#pragma unroll
        for (int r = 0; r < 8; ++r) {
            int   m = tileM * 16 + (lane >> 4) * 8 + r;
            float v = acc[s][r] + bias;
            if (t == 0)      Q [((size_t)hh * SEQ + m) * HEAD_DIM + d] = (half_t)(v * SCALE_Q);
            else if (t == 1) K [((size_t)hh * SEQ + m) * HEAD_DIM + d] = (half_t)v;
            else             Vt[((size_t)hh * HEAD_DIM + d) * SEQ + m] = (half_t)v;
        }
    }
}

// ---------------- Flash attention ----------------
// grid (SEQ/128, NHEAD), 256 threads. Each wave owns 16 query rows and streams
// keys in chunks of 64. All K and V fragments for a chunk are issued up front:
// the V loads remain in flight through the score WMMAs + softmax + transpose.

__global__ void tra_flash_attn(const half_t* __restrict__ Q, const half_t* __restrict__ K,
                               const half_t* __restrict__ Vt, half_t* __restrict__ AO) {
    __shared__ half_t ldsP[8][16][64];   // per-wave P tile (C-layout -> A-layout transpose)

    int lane = threadIdx.x & 31, wave = threadIdx.x >> 5;
    int h     = blockIdx.y;
    int qbase = blockIdx.x * 128 + wave * 16;

    const half_t* qrow = Q + ((size_t)h * SEQ + qbase + (lane & 15)) * HEAD_DIM;
    v16h qa0 = load_a_frag(qrow, lane);        // d  0..31
    v16h qa1 = load_a_frag(qrow + 32, lane);   // d 32..63

    const half_t* kbase = K  + (size_t)h * SEQ * HEAD_DIM;
    const half_t* vbase = Vt + (size_t)h * HEAD_DIM * SEQ;

    v8f o[4] = {{}, {}, {}, {}};
    float mrow[8], lrow[8];
#pragma unroll
    for (int r = 0; r < 8; ++r) { mrow[r] = -INFINITY; lrow[r] = 0.f; }

    for (int j0 = 0; j0 < SEQ; j0 += 64) {
        // prefetch next chunk's K (8 KB contiguous) and V (64 strided cachelines)
        if (j0 + 64 < SEQ) {
            __builtin_prefetch(kbase + (size_t)(j0 + 64) * HEAD_DIM + lane * 128, 0, 0);
            __builtin_prefetch(vbase + (size_t)lane        * SEQ + j0 + 64, 0, 0);
            __builtin_prefetch(vbase + (size_t)(32 + lane) * SEQ + j0 + 64, 0, 0);
        }

        // ---- issue ALL fragment loads for this chunk up front ----
        v16h kb[8], vb[8];
#pragma unroll
        for (int t = 0; t < 4; ++t) {
            const half_t* kp = kbase + (size_t)(j0 + t * 16 + (lane & 15)) * HEAD_DIM;
            kb[2 * t]     = load_b_frag(kp,      lane);
            kb[2 * t + 1] = load_b_frag(kp + 32, lane);
        }
#pragma unroll
        for (int t = 0; t < 4; ++t) {
            const half_t* vp = vbase + (size_t)(t * 16 + (lane & 15)) * SEQ + j0;
            vb[2 * t]     = load_b_frag(vp,      lane);
            vb[2 * t + 1] = load_b_frag(vp + 32, lane);
        }

        // ---- scores S = Q @ K^T (16x64, scale pre-folded into Q) ----
        v8f s[4];
        v8f z = {};
#pragma unroll
        for (int t = 0; t < 4; ++t) {
            v8f a = wmma_f16(qa0, kb[2 * t],     z);
            s[t]  = wmma_f16(qa1, kb[2 * t + 1], a);
        }

        // ---- online softmax: rows live in the VGPR index, keys across 16 lanes ----
        float alpha[8];
#pragma unroll
        for (int r = 0; r < 8; ++r) {
            float cm = fmaxf(fmaxf(s[0][r], s[1][r]), fmaxf(s[2][r], s[3][r]));
#pragma unroll
            for (int off = 1; off < 16; off <<= 1) cm = fmaxf(cm, __shfl_xor(cm, off, 32));
            float mn = fmaxf(mrow[r], cm);
            float a  = __expf(mrow[r] - mn);
            float p0 = __expf(s[0][r] - mn);
            float p1 = __expf(s[1][r] - mn);
            float p2 = __expf(s[2][r] - mn);
            float p3 = __expf(s[3][r] - mn);
            float ps = (p0 + p1) + (p2 + p3);
#pragma unroll
            for (int off = 1; off < 16; off <<= 1) ps += __shfl_xor(ps, off, 32);
            lrow[r]  = lrow[r] * a + ps;
            mrow[r]  = mn;
            alpha[r] = a;
            s[0][r] = p0; s[1][r] = p1; s[2][r] = p2; s[3][r] = p3;
        }
#pragma unroll
        for (int t = 0; t < 4; ++t)
#pragma unroll
            for (int r = 0; r < 8; ++r) o[t][r] *= alpha[r];

        // ---- P: C-layout -> LDS -> A-layout (wave-private tile; same-wave DS in order) ----
        int nloc = lane & 15, hs = (lane >> 4) * 8;
#pragma unroll
        for (int r = 0; r < 8; ++r)
#pragma unroll
            for (int t = 0; t < 4; ++t)
                ldsP[wave][hs + r][t * 16 + nloc] = (half_t)s[t][r];
        asm volatile("s_wait_dscnt 0" ::: "memory");
        v16h pa0 = load_a_frag(&ldsP[wave][lane & 15][0],  lane);  // keys j0   .. j0+31
        v16h pa1 = load_a_frag(&ldsP[wave][lane & 15][32], lane);  // keys j0+32.. j0+63

        // ---- O += P @ V  (V frags already resident) ----
#pragma unroll
        for (int t = 0; t < 4; ++t) {
            o[t] = wmma_f16(pa0, vb[2 * t],     o[t]);
            o[t] = wmma_f16(pa1, vb[2 * t + 1], o[t]);
        }
    }

    // epilogue: out = (O / l) * HEAD_DIM, stored f16 row-major [SEQ][D_MODEL]
    int nloc = lane & 15;
#pragma unroll
    for (int r = 0; r < 8; ++r) {
        int   m   = qbase + (lane >> 4) * 8 + r;
        float inv = 64.0f / lrow[r];
        half_t* dst = AO + (size_t)m * D_MODEL + h * HEAD_DIM + nloc;
#pragma unroll
        for (int t = 0; t < 4; ++t) dst[t * 16] = (half_t)(o[t][r] * inv);
    }
}

// ---------------- O GEMM: AO[4096][768] @ WoT[768][768]^T + b_o -> f32 ----------------
// One wave computes a 16x64 output strip, software-pipelined like the QKV GEMM.

__global__ void tra_o_gemm(const half_t* __restrict__ AO, const half_t* __restrict__ WoT,
                           const float* __restrict__ bo, float* __restrict__ out) {
    int lane = threadIdx.x & 31, wave = threadIdx.x >> 5;
    int tile  = blockIdx.x * 8 + wave;
    int tileN = tile % (D_MODEL / 64);
    int tileM = tile / (D_MODEL / 64);

    const half_t* arow = AO  + (size_t)(tileM * 16 + (lane & 15)) * D_MODEL;
    const half_t* brow = WoT + (size_t)(tileN * 64 + (lane & 15)) * D_MODEL;

    v8f acc[4] = {{}, {}, {}, {}};

    v16h a0 = load_a_frag(arow, lane);
    v16h b0[4];
#pragma unroll
    for (int s = 0; s < 4; ++s) b0[s] = load_b_frag(brow + (size_t)s * 16 * D_MODEL, lane);

#pragma unroll 2
    for (int k0 = 0; k0 < D_MODEL; k0 += 32) {
        int k1 = (k0 + 32 == D_MODEL) ? 0 : k0 + 32;   // branchless tail (redundant reload)
        v16h a1 = load_a_frag(arow + k1, lane);
        v16h b1[4];
#pragma unroll
        for (int s = 0; s < 4; ++s)
            b1[s] = load_b_frag(brow + (size_t)s * 16 * D_MODEL + k1, lane);
#pragma unroll
        for (int s = 0; s < 4; ++s) acc[s] = wmma_f16(a0, b0[s], acc[s]);
        a0 = a1;
#pragma unroll
        for (int s = 0; s < 4; ++s) b0[s] = b1[s];
    }

#pragma unroll
    for (int s = 0; s < 4; ++s) {
        int   n = tileN * 64 + s * 16 + (lane & 15);
        float b = bo[n];
#pragma unroll
        for (int r = 0; r < 8; ++r) {
            int m = tileM * 16 + (lane >> 4) * 8 + r;
            out[(size_t)m * D_MODEL + n] = acc[s][r] + b;
        }
    }
}

// ---------------- launch ----------------

extern "C" void kernel_launch(void* const* d_in, const int* in_sizes, int n_in,
                              void* d_out, int out_size, void* d_ws, size_t ws_size,
                              hipStream_t stream) {
    const float* x    = (const float*)d_in[0];
    const float* Wqkv = (const float*)d_in[1];
    const float* bqkv = (const float*)d_in[2];
    const float* Wo   = (const float*)d_in[3];
    const float* bo   = (const float*)d_in[4];
    float* out = (float*)d_out;

    char* ws = (char*)d_ws;
    const size_t XN   = (size_t)SEQ * D_MODEL;          // 3,145,728
    const size_t WQN  = (size_t)D_MODEL * NQKV;         // 1,769,472
    const size_t WON  = (size_t)D_MODEL * D_MODEL;      //   589,824
    const size_t HQN  = (size_t)NHEAD * SEQ * HEAD_DIM; // = XN

    half_t* x16    = (half_t*)ws;                 ws += XN  * sizeof(half_t);
    half_t* WqkvT  = (half_t*)ws;                 ws += WQN * sizeof(half_t);
    half_t* WoT    = (half_t*)ws;                 ws += WON * sizeof(half_t);
    half_t* Qh     = (half_t*)ws;                 ws += HQN * sizeof(half_t);
    half_t* Kh     = (half_t*)ws;                 ws += HQN * sizeof(half_t);
    half_t* Vth    = (half_t*)ws;                 ws += HQN * sizeof(half_t);
    half_t* AO     = (half_t*)ws;                 ws += XN  * sizeof(half_t);

    tra_cvt_f32_to_f16<<<(int)((XN + 255) / 256), 256, 0, stream>>>(x, x16, (int)XN);
    tra_cvt_transpose <<<(int)((WQN + 255) / 256), 256, 0, stream>>>(Wqkv, WqkvT, D_MODEL, NQKV);
    tra_cvt_transpose <<<(int)((WON + 255) / 256), 256, 0, stream>>>(Wo, WoT, D_MODEL, D_MODEL);

    // (4096/16) * (2304/64) strips / 8 waves per block
    tra_qkv_gemm<<<(SEQ / 16) * (NQKV / 64) / 8, 256, 0, stream>>>(x16, WqkvT, bqkv, Qh, Kh, Vth);

    tra_flash_attn<<<dim3(SEQ / 128, NHEAD), 256, 0, stream>>>(Qh, Kh, Vth, AO);

    tra_o_gemm<<<(SEQ / 16) * (D_MODEL / 64) / 8, 256, 0, stream>>>(AO, WoT, bo, out);
}
